// GraphGenRnn_63780264345608
// MI455X (gfx1250) — matrix operand order repaired and checked by reading, not compiled, not run on hardware
//
#include <hip/hip_runtime.h>
#include <hip/hip_bf16.h>

// ---------------------------------------------------------------------------
// GraphGenRnn for MI455X (gfx1250): all matmuls via v_wmma_f32_16x16x32_f16.
// ---------------------------------------------------------------------------

typedef __attribute__((ext_vector_type(16))) _Float16 v16h;
typedef __attribute__((ext_vector_type(8)))  _Float16 v8h;
typedef __attribute__((ext_vector_type(8)))  float    v8f;

#define S_LEN 160
#define G_N   64
#define NMAX  161
#define M_TOT (S_LEN * G_N)   // 10240

__device__ __forceinline__ float sigf(float x) { return 1.0f / (1.0f + __expf(-x)); }

__device__ __forceinline__ v8f splat8(float x) {
    v8f r;
#pragma unroll
    for (int i = 0; i < 8; ++i) r[i] = x;
    return r;
}

// WMMA f32 += f16 A[16x32] * f16 B[32x16]
__device__ __forceinline__ v8f wmma16(v16h a, v16h b, v8f c) {
    return __builtin_amdgcn_wmma_f32_16x16x32_f16(false, a, false, b, (short)0, c,
                                                  false, false);
}

// A-fragment (16x32, f16) from a row-major f16 matrix (generic ptr: LDS or global).
// ISA layout: lanes 0-15 hold K 0-7 / 16-23; lanes 16-31 hold K 8-15 / 24-31; M=lane%16.
__device__ __forceinline__ v16h load_a16(const _Float16* A, int lda, int row0, int col0) {
    int lane = threadIdx.x & 31;
    const _Float16* p = A + (row0 + (lane & 15)) * lda + col0;
    int kb = (lane < 16) ? 0 : 8;
    v8h lo = *(const v8h*)(p + kb);
    v8h hi = *(const v8h*)(p + kb + 16);
    v16h r;
#pragma unroll
    for (int i = 0; i < 8; ++i) { r[i] = lo[i]; r[8 + i] = hi[i]; }
    return r;
}

// B-fragment from pre-packed fragment storage: one contiguous 32B load per lane.
// Packed layout: dst[(((kt*NT)+nt)*32 + lane)*16 + i]
__device__ __forceinline__ v16h load_b_pk(const _Float16* Pk, int NT, int kt, int nt) {
    int lane = threadIdx.x & 31;
    return *(const v16h*)(Pk + ((((kt * NT) + nt) * 32) + lane) * 16);
}

// ---------------------------------------------------------------------------
// Utility kernels
// ---------------------------------------------------------------------------
__global__ void zero_kernel(float* p, long n) {
    long i = (long)blockIdx.x * blockDim.x + threadIdx.x;
    long st = (long)gridDim.x * blockDim.x;
    for (; i < n; i += st) p[i] = 0.0f;
}

// Pack W (Nout x Kin, row-major f32, PyTorch layout) as WMMA-B fragments of W^T
// (K x N), K zero-padded to KT*32.
__global__ void pack_kernel(const float* __restrict__ w, _Float16* __restrict__ dst,
                            int Nout, int Kin, int KT, int NT) {
    int total = KT * NT * 512;
    for (int idx = blockIdx.x * blockDim.x + threadIdx.x; idx < total;
         idx += gridDim.x * blockDim.x) {
        int i = idx & 15;
        int lane = (idx >> 4) & 31;
        int t = idx >> 9;
        int nt = t % NT;
        int kt = t / NT;
        int n = nt * 16 + (lane & 15);
        int k = kt * 32 + ((lane < 16) ? 0 : 8) + i + ((i < 8) ? 0 : 8);
        float v = (k < Kin && n < Nout) ? w[n * Kin + k] : 0.0f;
        dst[idx] = (_Float16)v;
    }
}

// f16 copy of gene codes + combined biases (bih + bhh).
__global__ void misc_convert_kernel(const float* __restrict__ gene, _Float16* __restrict__ gene16,
                                    const float* __restrict__ nbih, const float* __restrict__ nbhh,
                                    float* __restrict__ nb0, float* __restrict__ nb1,
                                    const float* __restrict__ ebih, const float* __restrict__ ebhh,
                                    float* __restrict__ eb0, float* __restrict__ eb1) {
    int tid = blockIdx.x * blockDim.x + threadIdx.x;
    int st = gridDim.x * blockDim.x;
    for (int i = tid; i < 64 * 128; i += st) gene16[i] = (_Float16)gene[i];
    for (int i = tid; i < 512; i += st) {
        nb0[i] = nbih[i] + nbhh[i];
        nb1[i] = nbih[512 + i] + nbhh[512 + i];
    }
    for (int i = tid; i < 64; i += st) {
        eb0[i] = ebih[i] + ebhh[i];
        eb1[i] = ebih[64 + i] + ebhh[64 + i];
    }
}

// ---------------------------------------------------------------------------
// h0n = relu(gene @ node_entry_w^T + b)  [64x128], one block, 8 waves x 4 tiles
// ---------------------------------------------------------------------------
__global__ __launch_bounds__(256) void h0n_kernel(const _Float16* __restrict__ gene16,
                                                  const _Float16* __restrict__ entryPk,
                                                  const float* __restrict__ bvec,
                                                  _Float16* __restrict__ h0n16) {
    int wave = threadIdx.x >> 5, lane = threadIdx.x & 31;
    int nl = lane & 15, mb = (lane < 16) ? 0 : 8;
    for (int t = wave; t < 32; t += 8) {
        int m = t & 3, n = t >> 2;
        v8f acc = splat8(bvec[n * 16 + nl]);
#pragma unroll
        for (int kc = 0; kc < 4; ++kc)
            acc = wmma16(load_a16(gene16, 128, m * 16, kc * 32),
                         load_b_pk(entryPk, 8, kc, n), acc);
#pragma unroll
        for (int r = 0; r < 8; ++r) {
            float v = acc[r];
            h0n16[(m * 16 + mb + r) * 128 + n * 16 + nl] = (_Float16)(v > 0.f ? v : 0.f);
        }
    }
}

// ---------------------------------------------------------------------------
// Node LSTM: 1 block, 1024 threads = 32 waves = 4 batch-tiles x 8 hidden-tiles.
// Each wave keeps its c0/c1 fragments in registers; i/f/g/o accumulators share
// lane mapping so pointwise LSTM is lane-local. Writes top hidden in flipped
// packed order (f16) for the edge stage.
// ---------------------------------------------------------------------------
__global__ __launch_bounds__(1024) void node_lstm_kernel(
    const _Float16* __restrict__ h0n16,
    const _Float16* __restrict__ whh0Pk, const _Float16* __restrict__ wihPk,
    const _Float16* __restrict__ whh1Pk,
    const float* __restrict__ bias0, const float* __restrict__ bias1,
    _Float16* __restrict__ nodeFlat) {
    __shared__ __align__(16) _Float16 h0[64 * 128];
    __shared__ __align__(16) _Float16 h1[64 * 128];
    int tid = threadIdx.x;
    for (int i = tid; i < 64 * 128; i += 1024) {
        _Float16 v = h0n16[i];
        h0[i] = v;
        h1[i] = v;
    }
    __syncthreads();

    int wave = tid >> 5, lane = tid & 31;
    int m = wave & 3;    // batch tile
    int j = wave >> 2;   // hidden tile
    int nl = lane & 15, mb = (lane < 16) ? 0 : 8;

    v8f c0 = splat8(0.f), c1 = splat8(0.f);
    float b0g[4], b1g[4];
#pragma unroll
    for (int g = 0; g < 4; ++g) {
        b0g[g] = bias0[g * 128 + j * 16 + nl];
        b1g[g] = bias1[g * 128 + j * 16 + nl];
    }
    const int NT = 32;  // 512/16 N-tiles in the gate matrices

    for (int step = 0; step < S_LEN; ++step) {
        __builtin_prefetch(whh0Pk + ((j * 32 + lane) * 16), 0, 0);
        // ---- layer 0: gates = h0 @ Whh0^T + b0 ----
        v8f acc[4];
#pragma unroll
        for (int g = 0; g < 4; ++g) acc[g] = splat8(b0g[g]);
#pragma unroll
        for (int kc = 0; kc < 4; ++kc) {
            v16h a = load_a16(h0, 128, m * 16, kc * 32);
#pragma unroll
            for (int g = 0; g < 4; ++g)
                acc[g] = wmma16(a, load_b_pk(whh0Pk, NT, kc, g * 8 + j), acc[g]);
        }
        __syncthreads();  // all reads of h0 done
#pragma unroll
        for (int r = 0; r < 8; ++r) {
            float ci = sigf(acc[1][r]) * c0[r] + sigf(acc[0][r]) * tanhf(acc[2][r]);
            c0[r] = ci;
            float hh = sigf(acc[3][r]) * tanhf(ci);
            h0[(m * 16 + mb + r) * 128 + j * 16 + nl] = (_Float16)hh;
        }
        __syncthreads();  // new h0 visible
        // ---- layer 1: gates = h0_new @ Wih^T + h1 @ Whh1^T + b1 ----
#pragma unroll
        for (int g = 0; g < 4; ++g) acc[g] = splat8(b1g[g]);
#pragma unroll
        for (int kc = 0; kc < 4; ++kc) {
            v16h a0 = load_a16(h0, 128, m * 16, kc * 32);
            v16h a1 = load_a16(h1, 128, m * 16, kc * 32);
#pragma unroll
            for (int g = 0; g < 4; ++g) {
                acc[g] = wmma16(a0, load_b_pk(wihPk, NT, kc, g * 8 + j), acc[g]);
                acc[g] = wmma16(a1, load_b_pk(whh1Pk, NT, kc, g * 8 + j), acc[g]);
            }
        }
        __syncthreads();  // all reads of h1 done
        int rowbase = (S_LEN - 1 - step) * G_N;  // flipped packed layout
#pragma unroll
        for (int r = 0; r < 8; ++r) {
            float ci = sigf(acc[1][r]) * c1[r] + sigf(acc[0][r]) * tanhf(acc[2][r]);
            c1[r] = ci;
            _Float16 hh = (_Float16)(sigf(acc[3][r]) * tanhf(ci));
            h1[(m * 16 + mb + r) * 128 + j * 16 + nl] = hh;
            nodeFlat[(rowbase + m * 16 + mb + r) * 128 + j * 16 + nl] = hh;
        }
        __syncthreads();
    }
}

// ---------------------------------------------------------------------------
// h0e = relu(relu(nodeFlat @ exitW^T + b) @ edge_entryW^T + b)  [10240 x 16]
// stored K-padded to 32 (zeros) so the edge kernel's A-frags are uniform.
// ---------------------------------------------------------------------------
__global__ __launch_bounds__(256) void h0e_kernel(const _Float16* __restrict__ nodeFlat,
                                                  const _Float16* __restrict__ exitPk,
                                                  const float* __restrict__ exit_b,
                                                  const _Float16* __restrict__ entryEPk,
                                                  const float* __restrict__ entry_b,
                                                  _Float16* __restrict__ h0e16) {
    __shared__ __align__(16) _Float16 t1[64 * 32];
    int rb = blockIdx.x * 64;
    int wave = threadIdx.x >> 5, lane = threadIdx.x & 31;
    int nl = lane & 15, mb = (lane < 16) ? 0 : 8;
    int m = wave & 3, n = wave >> 2;  // [64x32]: 4 M-tiles x 2 N-tiles

    v8f acc = splat8(exit_b[n * 16 + nl]);
#pragma unroll
    for (int kc = 0; kc < 4; ++kc)
        acc = wmma16(load_a16(nodeFlat + (long)rb * 128, 128, m * 16, kc * 32),
                     load_b_pk(exitPk, 2, kc, n), acc);
#pragma unroll
    for (int r = 0; r < 8; ++r) {
        float v = acc[r];
        t1[(m * 16 + mb + r) * 32 + n * 16 + nl] = (_Float16)(v > 0.f ? v : 0.f);
    }
    __syncthreads();
    if (wave < 4) {  // [64x32] @ [32x16]
        v8f a2 = splat8(entry_b[nl]);
        a2 = wmma16(load_a16(t1, 32, wave * 16, 0), load_b_pk(entryEPk, 1, 0, 0), a2);
#pragma unroll
        for (int r = 0; r < 8; ++r) {
            float v = a2[r];
            int rr = rb + wave * 16 + mb + r;
            h0e16[rr * 32 + nl] = (_Float16)(v > 0.f ? v : 0.f);
            h0e16[rr * 32 + 16 + nl] = (_Float16)0.f;  // K padding
        }
    }
}

// ---------------------------------------------------------------------------
// Edge LSTM + heads + scatter. Block b owns sequences k in [64b, 64b+64):
// all share row = 160 - b (so the block runs only `row` steps) and gid = k%64
// = local index. Scatter positions are collision-free -> plain stores.
// Waves 0-3: layer 0. Waves 4-7: layer 1 (start with h1@Whh1 concurrently).
// ---------------------------------------------------------------------------
#define EE1W 0
#define EE1B 128
#define EE2W 136
#define EE2B 152
#define EF1W 160
#define EF1B 288
#define EF2W 296
#define EF2B 552

__global__ __launch_bounds__(256) void edge_lstm_kernel(
    const _Float16* __restrict__ h0e16,
    const _Float16* __restrict__ whh0Pk, const _Float16* __restrict__ wihPk,
    const _Float16* __restrict__ whh1Pk,
    const float* __restrict__ bias0, const float* __restrict__ bias1,
    const float* __restrict__ ee1_w, const float* __restrict__ ee1_b,
    const float* __restrict__ ee2_w, const float* __restrict__ ee2_b,
    const float* __restrict__ ef1_w, const float* __restrict__ ef1_b,
    const float* __restrict__ ef2_w, const float* __restrict__ ef2_b,
    float* __restrict__ exist, float* __restrict__ feat) {
    __shared__ __align__(16) _Float16 h0[64 * 32];
    __shared__ __align__(16) _Float16 h1[64 * 32];
    __shared__ float eh[64 * 16];
    __shared__ __align__(32) _Float16 wf[3 * 2048];  // packed B-frags of 3 gate matrices
    __shared__ float hw[584];                        // head weights

    int b = blockIdx.x;
    int row = S_LEN - b;  // uniform edge length in this block
    int tid = threadIdx.x;

    for (int i = tid; i < 64 * 32; i += 256) {
        _Float16 v = h0e16[(long)b * 2048 + i];
        h0[i] = v;
        h1[i] = v;
    }
    for (int i = tid; i < 2048; i += 256) {
        wf[i] = whh0Pk[i];
        wf[2048 + i] = wihPk[i];
        wf[4096 + i] = whh1Pk[i];
    }
    for (int i = tid; i < 128; i += 256) { hw[EE1W + i] = ee1_w[i]; hw[EF1W + i] = ef1_w[i]; }
    for (int i = tid; i < 8; i += 256)   { hw[EE1B + i] = ee1_b[i]; hw[EF1B + i] = ef1_b[i]; }
    for (int i = tid; i < 16; i += 256)  hw[EE2W + i] = ee2_w[i];
    for (int i = tid; i < 2; i += 256)   hw[EE2B + i] = ee2_b[i];
    for (int i = tid; i < 256; i += 256) hw[EF2W + i] = ef2_w[i];
    for (int i = tid; i < 32; i += 256)  hw[EF2B + i] = ef2_b[i];
    __syncthreads();

    int wave = tid >> 5, lane = tid & 31;
    int m = wave & 3;        // batch tile (16 sequences)
    bool grpA = wave < 4;    // layer-0 group
    int nl = lane & 15, mb = (lane < 16) ? 0 : 8;
    v8f cfrag = splat8(0.f);
    float bg[4];
#pragma unroll
    for (int g = 0; g < 4; ++g) bg[g] = (grpA ? bias0 : bias1)[g * 16 + nl];

    for (int s = 0; s < row; ++s) {
        v8f acc[4];
#pragma unroll
        for (int g = 0; g < 4; ++g) acc[g] = splat8(bg[g]);
        if (grpA) {
            v16h a = load_a16(h0, 32, m * 16, 0);
#pragma unroll
            for (int g = 0; g < 4; ++g) acc[g] = wmma16(a, load_b_pk(wf, 4, 0, g), acc[g]);
        } else {
            v16h a1 = load_a16(h1, 32, m * 16, 0);
#pragma unroll
            for (int g = 0; g < 4; ++g) acc[g] = wmma16(a1, load_b_pk(wf + 4096, 4, 0, g), acc[g]);
        }
        __syncthreads();
        if (grpA) {  // layer-0 pointwise, write new h0
#pragma unroll
            for (int r = 0; r < 8; ++r) {
                float ci = sigf(acc[1][r]) * cfrag[r] + sigf(acc[0][r]) * tanhf(acc[2][r]);
                cfrag[r] = ci;
                h0[(m * 16 + mb + r) * 32 + nl] = (_Float16)(sigf(acc[3][r]) * tanhf(ci));
            }
        }
        __syncthreads();  // new h0 visible to layer-1 waves
        if (!grpA) {
            v16h a0 = load_a16(h0, 32, m * 16, 0);
#pragma unroll
            for (int g = 0; g < 4; ++g) acc[g] = wmma16(a0, load_b_pk(wf + 2048, 4, 0, g), acc[g]);
#pragma unroll
            for (int r = 0; r < 8; ++r) {
                float ci = sigf(acc[1][r]) * cfrag[r] + sigf(acc[0][r]) * tanhf(acc[2][r]);
                cfrag[r] = ci;
                float hh = sigf(acc[3][r]) * tanhf(ci);
                h1[(m * 16 + mb + r) * 32 + nl] = (_Float16)hh;
                eh[(m * 16 + mb + r) * 16 + nl] = hh;
            }
        }
        __syncthreads();  // eh ready for heads

        // ---- heads: 4 threads per sequence ----
        int part = tid & 3, seq = tid >> 2;  // seq == gid
        const float* ehs = &eh[seq * 16];
        float t1[8];
        if (part == 0) {  // existence head (both directions)
#pragma unroll
            for (int u = 0; u < 8; ++u) {
                float a2 = hw[EE1B + u];
#pragma unroll
                for (int k2 = 0; k2 < 16; ++k2) a2 += hw[EE1W + u * 16 + k2] * ehs[k2];
                t1[u] = a2 > 0.f ? a2 : 0.f;
            }
            float e0 = hw[EE2B + 0], e1 = hw[EE2B + 1];
#pragma unroll
            for (int u = 0; u < 8; ++u) {
                e0 += hw[EE2W + u] * t1[u];
                e1 += hw[EE2W + 8 + u] * t1[u];
            }
            exist[((long)seq * NMAX + row) * NMAX + s] = sigf(e0);
            exist[((long)seq * NMAX + s) * NMAX + row] = sigf(e1);
        } else if (part <= 2) {  // feature head, lower (part 1) / upper (part 2)
#pragma unroll
            for (int u = 0; u < 8; ++u) {
                float a2 = hw[EF1B + u];
#pragma unroll
                for (int k2 = 0; k2 < 16; ++k2) a2 += hw[EF1W + u * 16 + k2] * ehs[k2];
                t1[u] = a2 > 0.f ? a2 : 0.f;
            }
            int half = part - 1;
            int r1 = half ? s : row, c1 = half ? row : s;
            float* fb = feat + (((long)seq * NMAX + r1) * NMAX + c1) * 16;
#pragma unroll
            for (int ff = 0; ff < 16; ++ff) {
                float a2 = hw[EF2B + half * 16 + ff];
                const float* wrow = &hw[EF2W + (half * 16 + ff) * 8];
#pragma unroll
                for (int u = 0; u < 8; ++u) a2 += wrow[u] * t1[u];
                fb[ff] = a2;
            }
        }
        __syncthreads();  // heads done before eh is overwritten
    }
}

// ---------------------------------------------------------------------------
// Host launch
// ---------------------------------------------------------------------------
extern "C" void kernel_launch(void* const* d_in, const int* in_sizes, int n_in,
                              void* d_out, int out_size, void* d_ws, size_t ws_size,
                              hipStream_t stream) {
    const float* gene     = (const float*)d_in[0];
    const float* nentry_w = (const float*)d_in[1];
    const float* nentry_b = (const float*)d_in[2];
    const float* nwih     = (const float*)d_in[4];   // (1,512,128)
    const float* nwhh     = (const float*)d_in[5];   // (2,512,128)
    const float* nbih     = (const float*)d_in[6];
    const float* nbhh     = (const float*)d_in[7];
    const float* nexit_w  = (const float*)d_in[8];
    const float* nexit_b  = (const float*)d_in[9];
    const float* eentry_w = (const float*)d_in[10];
    const float* eentry_b = (const float*)d_in[11];
    const float* ewih     = (const float*)d_in[13];  // (1,64,16)
    const float* ewhh     = (const float*)d_in[14];  // (2,64,16)
    const float* ebih     = (const float*)d_in[15];
    const float* ebhh     = (const float*)d_in[16];
    const float* ee1_w = (const float*)d_in[17];
    const float* ee1_b = (const float*)d_in[18];
    const float* ee2_w = (const float*)d_in[19];
    const float* ee2_b = (const float*)d_in[20];
    const float* ef1_w = (const float*)d_in[21];
    const float* ef1_b = (const float*)d_in[22];
    const float* ef2_w = (const float*)d_in[23];
    const float* ef2_b = (const float*)d_in[24];

    char* ws = (char*)d_ws;
    _Float16* gene16    = (_Float16*)(ws + 0);         // 16384 B
    _Float16* entryPk   = (_Float16*)(ws + 16384);     // 32768 B
    _Float16* nwhh0Pk   = (_Float16*)(ws + 49152);     // 131072 B
    _Float16* nwihPk    = (_Float16*)(ws + 180224);    // 131072 B
    _Float16* nwhh1Pk   = (_Float16*)(ws + 311296);    // 131072 B
    _Float16* exitPk    = (_Float16*)(ws + 442368);    // 8192 B
    _Float16* eentryPk  = (_Float16*)(ws + 450560);    // 1024 B
    _Float16* ewhh0Pk   = (_Float16*)(ws + 451584);    // 4096 B
    _Float16* ewihPk    = (_Float16*)(ws + 455680);    // 4096 B
    _Float16* ewhh1Pk   = (_Float16*)(ws + 459776);    // 4096 B
    float*    nb0       = (float*)(ws + 463872);       // 2048 B
    float*    nb1       = (float*)(ws + 465920);       // 2048 B
    float*    eb0       = (float*)(ws + 467968);       // 256 B
    float*    eb1       = (float*)(ws + 468224);       // 256 B
    _Float16* h0n16     = (_Float16*)(ws + 468480);    // 16384 B
    _Float16* nodeFlat  = (_Float16*)(ws + 484864);    // 2621440 B
    _Float16* h0e16     = (_Float16*)(ws + 3106304);   // 655360 B

    float* exist = (float*)d_out;
    float* feat  = exist + (long)G_N * NMAX * NMAX;

    zero_kernel<<<4096, 256, 0, stream>>>((float*)d_out, (long)out_size);
    misc_convert_kernel<<<32, 256, 0, stream>>>(gene, gene16, nbih, nbhh, nb0, nb1,
                                                ebih, ebhh, eb0, eb1);
    pack_kernel<<<64, 256, 0, stream>>>(nentry_w, entryPk, 128, 128, 4, 8);
    pack_kernel<<<128, 256, 0, stream>>>(nwhh, nwhh0Pk, 512, 128, 4, 32);
    pack_kernel<<<128, 256, 0, stream>>>(nwih, nwihPk, 512, 128, 4, 32);
    pack_kernel<<<128, 256, 0, stream>>>(nwhh + 512 * 128, nwhh1Pk, 512, 128, 4, 32);
    pack_kernel<<<16, 256, 0, stream>>>(nexit_w, exitPk, 32, 128, 4, 2);
    pack_kernel<<<2, 256, 0, stream>>>(eentry_w, eentryPk, 16, 32, 1, 1);
    pack_kernel<<<8, 256, 0, stream>>>(ewhh, ewhh0Pk, 64, 16, 1, 4);
    pack_kernel<<<8, 256, 0, stream>>>(ewih, ewihPk, 64, 16, 1, 4);
    pack_kernel<<<8, 256, 0, stream>>>(ewhh + 64 * 16, ewhh1Pk, 64, 16, 1, 4);

    h0n_kernel<<<1, 256, 0, stream>>>(gene16, entryPk, nentry_b, h0n16);
    node_lstm_kernel<<<1, 1024, 0, stream>>>(h0n16, nwhh0Pk, nwihPk, nwhh1Pk,
                                             nb0, nb1, nodeFlat);
    h0e_kernel<<<M_TOT / 64, 256, 0, stream>>>(nodeFlat, exitPk, nexit_b,
                                               eentryPk, eentry_b, h0e16);
    edge_lstm_kernel<<<S_LEN, 256, 0, stream>>>(h0e16, ewhh0Pk, ewihPk, ewhh1Pk,
                                                eb0, eb1, ee1_w, ee1_b, ee2_w, ee2_b,
                                                ef1_w, ef1_b, ef2_w, ef2_b, exist, feat);
    (void)in_sizes; (void)n_in; (void)ws_size;
}
